// GatedDeltaNet_67774583931207
// MI455X (gfx1250) — matrix operand order repaired and checked by reading, not compiled, not run on hardware
//
#include <hip/hip_runtime.h>
#include <hip/hip_bf16.h>
#include <cstdint>

// ---------------------------------------------------------------------------
// GatedDeltaNet forward for gfx1250 (MI455X).  B=2, S=2048, DIM=2048, H=16,
// DK=DV=128, KCONV=4.  bf16 WMMA GEMMs with double-buffered async global->LDS
// pipeline + register-resident delta-rule scan.
// ---------------------------------------------------------------------------

#define BDIM   2048
#define SLEN   2048
#define NH     16
#define HD     128
#define MROWS  4096          // B*S
#define EPSNRM 1e-6f

typedef __attribute__((ext_vector_type(16))) __bf16 v16bf;
typedef __attribute__((ext_vector_type(8)))  float  v8f;
typedef int v4i_vs __attribute__((vector_size(16)));   // matches builtin proto

#if defined(__has_builtin)
#if __has_builtin(__builtin_amdgcn_global_load_async_to_lds_b128) && \
    __has_builtin(__builtin_amdgcn_s_wait_asynccnt)
#define USE_ASYNC_LDS 1
#endif
#endif

#define AS1 __attribute__((address_space(1)))
#define AS3 __attribute__((address_space(3)))

__device__ __forceinline__ float sigmoid_f(float x) { return 1.0f / (1.0f + expf(-x)); }
__device__ __forceinline__ float silu_f(float x)    { return x * sigmoid_f(x); }

// ------------------------------ f32 -> bf16 cast ---------------------------
__global__ void k_cast_bf16(const float* __restrict__ src, __bf16* __restrict__ dst, int n) {
    int i = blockIdx.x * blockDim.x + threadIdx.x;
    if (i < n) dst[i] = (__bf16)src[i];
}

// ------------------------------ bf16 WMMA GEMM -----------------------------
// C[m,n] = sum_k A[m,k] * Bm[n,k]     (i.e. C = A @ Bm^T)
// A: MxK bf16 row-major, Bm: NxK bf16 row-major, C: MxN f32.
// Block tile 64x256, K-step 32, 8 waves (wave32): wave (wm2 in 0..1, wn2 in
// 0..3) computes a 32x64 slice with eight v_wmma_f32_16x16x32_bf16.
// Double-buffered LDS: async loads of tile i+1 overlap WMMAs of tile i.
#define GBM 64
#define GBN 256
#define GBK 32

__global__ __launch_bounds__(256) void k_gemm_bf16(
    const __bf16* __restrict__ A, const __bf16* __restrict__ Bm,
    float* __restrict__ C, int M, int N, int K)
{
    __shared__ __bf16 As[2][GBM * GBK];   // 2 x 4 KB
    __shared__ __bf16 Bs[2][GBN * GBK];   // 2 x 16 KB

    const int tid  = threadIdx.x;
    const int lane = tid & 31;
    const int wave = tid >> 5;
    const int wm2  = wave & 1;   // 32-row slice within block tile
    const int wn2  = wave >> 1;  // 64-col slice within block tile
    const int m0   = blockIdx.y * GBM;
    const int n0   = blockIdx.x * GBN;

    // cooperative tile load mapping: 256 threads, 16B granules
    const int lrow = tid >> 2;        // 0..63
    const int lk   = (tid & 3) * 8;   // element offset 0,8,16,24

    auto issue = [&](int k0, int buf) {
        const __bf16* ga = &A[(size_t)(m0 + lrow) * K + k0 + lk];
#if defined(USE_ASYNC_LDS)
        __builtin_amdgcn_global_load_async_to_lds_b128(
            (AS1 v4i_vs*)ga, (AS3 v4i_vs*)&As[buf][lrow * GBK + lk], 0, 0);
#pragma unroll
        for (int r = 0; r < 4; ++r) {
            const __bf16* gb = &Bm[(size_t)(n0 + lrow + 64 * r) * K + k0 + lk];
            __builtin_amdgcn_global_load_async_to_lds_b128(
                (AS1 v4i_vs*)gb, (AS3 v4i_vs*)&Bs[buf][(lrow + 64 * r) * GBK + lk], 0, 0);
        }
#else
        *(uint4*)(&As[buf][lrow * GBK + lk]) = *(const uint4*)ga;
#pragma unroll
        for (int r = 0; r < 4; ++r) {
            const __bf16* gb = &Bm[(size_t)(n0 + lrow + 64 * r) * K + k0 + lk];
            *(uint4*)(&Bs[buf][(lrow + 64 * r) * GBK + lk]) = *(const uint4*)gb;
        }
#endif
    };

    v8f acc[2][4] = {};

    issue(0, 0);
#if defined(USE_ASYNC_LDS)
    __builtin_amdgcn_s_wait_asynccnt(0);
#endif
    __syncthreads();

    int buf = 0;
    for (int k0 = 0; k0 < K; k0 += GBK, buf ^= 1) {
        if (k0 + GBK < K) issue(k0 + GBK, buf ^ 1);   // prefetch next tile (async)

        // A fragments (16x32 bf16 each): lanes 0-15 row=lane K{0..7,16..23},
        //                                lanes 16-31 row=lane-16 K{8..15,24..31}
        const int ak = (lane >> 4) * 8;
        const int ar = wm2 * 32 + (lane & 15);
        v16bf a0, a1;
        {
            uint4* p0 = (uint4*)&a0;
            p0[0] = *(const uint4*)(&As[buf][ar * GBK + ak]);
            p0[1] = *(const uint4*)(&As[buf][ar * GBK + ak + 16]);
            uint4* p1 = (uint4*)&a1;
            p1[0] = *(const uint4*)(&As[buf][(ar + 16) * GBK + ak]);
            p1[1] = *(const uint4*)(&As[buf][(ar + 16) * GBK + ak + 16]);
        }
        // B fragments (32x16 bf16 each): lane n = column, lanes 0-15 K{0..7,16..23},
        //                                lanes 16-31 K{8..15,24..31}
        const int bk = (lane >> 4) * 8;
        const int bc = wn2 * 64 + (lane & 15);
        v16bf bf[4];
#pragma unroll
        for (int j = 0; j < 4; ++j) {
            uint4* pj = (uint4*)&bf[j];
            pj[0] = *(const uint4*)(&Bs[buf][(bc + 16 * j) * GBK + bk]);
            pj[1] = *(const uint4*)(&Bs[buf][(bc + 16 * j) * GBK + bk + 16]);
        }

#pragma unroll
        for (int j = 0; j < 4; ++j) {
            acc[0][j] = __builtin_amdgcn_wmma_f32_16x16x32_bf16(false, a0, false, bf[j],
                                                                (short)0, acc[0][j], false, false);
            acc[1][j] = __builtin_amdgcn_wmma_f32_16x16x32_bf16(false, a1, false, bf[j],
                                                                (short)0, acc[1][j], false, false);
        }

#if defined(USE_ASYNC_LDS)
        __builtin_amdgcn_s_wait_asynccnt(0);   // next tile landed
#endif
        __syncthreads();
    }

    // C layout: lane 0-15 -> N=lane, M=vgpr; lanes 16-31 -> M=8+vgpr
    const int cn = lane & 15;
    const int mb = (lane >> 4) * 8;
#pragma unroll
    for (int i = 0; i < 2; ++i) {
#pragma unroll
        for (int j = 0; j < 4; ++j) {
            const int row0 = m0 + wm2 * 32 + i * 16 + mb;
            const int col0 = n0 + wn2 * 64 + j * 16 + cn;
#pragma unroll
            for (int r = 0; r < 8; ++r)
                C[(size_t)(row0 + r) * N + col0] = acc[i][j][r];
        }
    }
}

// --------------------- small projections a,b (H=16 each) -------------------
__global__ __launch_bounds__(256) void k_proj_ab(
    const float* __restrict__ x, const float* __restrict__ Wa,
    const float* __restrict__ Wb, float* __restrict__ a, float* __restrict__ b)
{
    const int row  = blockIdx.x;          // 0..MROWS-1
    const int wave = threadIdx.x >> 5;
    const int lane = threadIdx.x & 31;
    const float* xr = x + (size_t)row * BDIM;
    for (int h = wave; h < NH; h += 8) {
        const float* wa = Wa + (size_t)h * BDIM;
        const float* wb = Wb + (size_t)h * BDIM;
        float sa = 0.f, sb = 0.f;
        for (int i = lane; i < BDIM; i += 32) {
            float xv = xr[i];
            sa = fmaf(xv, wa[i], sa);
            sb = fmaf(xv, wb[i], sb);
        }
        for (int off = 16; off; off >>= 1) {
            sa += __shfl_down(sa, off, 32);
            sb += __shfl_down(sb, off, 32);
        }
        if (lane == 0) {
            a[(size_t)row * NH + h] = sa;
            b[(size_t)row * NH + h] = sb;
        }
    }
}

// ------------------------------ gate / beta --------------------------------
__global__ void k_gate(const float* __restrict__ a, const float* __restrict__ bt,
                       const float* __restrict__ A_log, const float* __restrict__ dt_bias,
                       float* __restrict__ g, float* __restrict__ beta, int n)
{
    int i = blockIdx.x * blockDim.x + threadIdx.x;
    if (i >= n) return;
    int h = i & (NH - 1);
    float z  = a[i] + dt_bias[h];
    float sp = (z > 20.f) ? z : log1pf(expf(z));
    g[i]    = expf(-expf(A_log[h]) * sp);
    beta[i] = sigmoid_f(bt[i]);
}

// ---------------------- depthwise causal conv(4) + SiLU --------------------
__global__ void k_conv_silu(const float* __restrict__ in, const float* __restrict__ w,
                            const float* __restrict__ bias, float* __restrict__ out,
                            int total, int C)
{
    int i = blockIdx.x * blockDim.x + threadIdx.x;
    if (i >= total) return;
    int c = i % C;
    int s = (i / C) % SLEN;
    const float* wr = w + (size_t)c * 4;
    float acc = bias[c];
#pragma unroll
    for (int j = 0; j < 4; ++j) {
        int sp = s + j - 3;
        if (sp >= 0) acc = fmaf(in[(long)i + (long)(j - 3) * C], wr[j], acc);
    }
    out[i] = silu_f(acc);
}

// ---------------------------- per-head L2 norm -----------------------------
__global__ __launch_bounds__(256) void k_l2norm(float* __restrict__ q, int nvec)
{
    int vec  = blockIdx.x * 8 + (threadIdx.x >> 5);
    int lane = threadIdx.x & 31;
    if (vec >= nvec) return;
    float* p = q + (size_t)vec * HD;
    float ss = 0.f;
    for (int i = lane; i < HD; i += 32) { float v = p[i]; ss = fmaf(v, v, ss); }
    for (int off = 16; off; off >>= 1) ss += __shfl_down(ss, off, 32);
    ss = __shfl(ss, 0, 32);
    float scale = 1.0f / fmaxf(sqrtf(ss), 1e-12f);
    for (int i = lane; i < HD; i += 32) p[i] *= scale;
}

// ------------------- gated delta-rule recurrence (per b,h) -----------------
// One block per (b,h). 256 threads = 16(tk) x 16(tv); each thread owns an
// 8x8 sub-block of the 128x128 state in registers.
__global__ __launch_bounds__(256) void k_recurrence(
    const float* __restrict__ q, const float* __restrict__ k,
    const float* __restrict__ v, const float* __restrict__ g,
    const float* __restrict__ beta, float* __restrict__ o)
{
    const int bh = blockIdx.x;           // 0..B*H-1
    const int b  = bh >> 4;
    const int h  = bh & (NH - 1);
    const int tid = threadIdx.x;
    const int tk  = tid >> 4;            // 0..15
    const int tv  = tid & 15;            // 0..15

    __shared__ float ks[HD], qs[HD], vs[HD], retr[HD], oacc[HD];
    __shared__ float gb[2];

    float Sreg[8][8];
#pragma unroll
    for (int i = 0; i < 8; ++i)
#pragma unroll
        for (int j = 0; j < 8; ++j) Sreg[i][j] = 0.f;

    for (int s = 0; s < SLEN; ++s) {
        const size_t base = (((size_t)b * SLEN + s) * NH + h) * HD;
        const size_t sb   = ((size_t)b * SLEN + s) * NH + h;
        if (tid < HD) {
            ks[tid] = k[base + tid];
            qs[tid] = q[base + tid];
            vs[tid] = v[base + tid];
            retr[tid] = 0.f;
            oacc[tid] = 0.f;
        } else if (tid == HD) {
            gb[0] = g[sb];
            gb[1] = beta[sb];
        }
        __syncthreads();

        float kv[8], qv[8];
#pragma unroll
        for (int i = 0; i < 8; ++i) {
            kv[i] = ks[tk + 16 * i];
            qv[i] = qs[tk + 16 * i];
        }
        // retrieved[v] = sum_k S[k,v] * k[k]  (partial over owned k-rows)
#pragma unroll
        for (int j = 0; j < 8; ++j) {
            float p = 0.f;
#pragma unroll
            for (int i = 0; i < 8; ++i) p = fmaf(Sreg[i][j], kv[i], p);
            atomicAdd(&retr[tv + 16 * j], p);        // ds_add_f32
        }
        __syncthreads();

        const float gg = gb[0], bb = gb[1];
#pragma unroll
        for (int j = 0; j < 8; ++j) {
            const int vc = tv + 16 * j;
            const float u = bb * (vs[vc] - gg * retr[vc]);
            float op = 0.f;
#pragma unroll
            for (int i = 0; i < 8; ++i) {
                float sn = fmaf(gg, Sreg[i][j], kv[i] * u);
                Sreg[i][j] = sn;
                op = fmaf(sn, qv[i], op);
            }
            atomicAdd(&oacc[vc], op);
        }
        __syncthreads();

        if (tid < HD) o[base + tid] = oacc[tid];
        __syncthreads();
    }
}

// ----------------- RMSNorm * norm_w * silu(gproj) -> bf16 ------------------
__global__ __launch_bounds__(256) void k_norm_gate(
    const float* __restrict__ o, const float* __restrict__ gp,
    const float* __restrict__ norm_w, __bf16* __restrict__ y)
{
    const int row = blockIdx.x;
    __shared__ float red[8];
    const float* orow = o + (size_t)row * BDIM;
    float ss = 0.f;
    for (int i = threadIdx.x; i < BDIM; i += 256) {
        float v = orow[i];
        ss = fmaf(v, v, ss);
    }
    for (int off = 16; off; off >>= 1) ss += __shfl_down(ss, off, 32);
    if ((threadIdx.x & 31) == 0) red[threadIdx.x >> 5] = ss;
    __syncthreads();
    if (threadIdx.x < 8) {
        float t = red[threadIdx.x];
        for (int off = 4; off; off >>= 1) t += __shfl_down(t, off, 8);
        if (threadIdx.x == 0) red[0] = t;
    }
    __syncthreads();
    const float scale = rsqrtf(red[0] / (float)BDIM + EPSNRM);
    const float* gr = gp + (size_t)row * BDIM;
    __bf16* yr = y + (size_t)row * BDIM;
    for (int i = threadIdx.x; i < BDIM; i += 256) {
        float val = orow[i] * scale * norm_w[i] * silu_f(gr[i]);
        yr[i] = (__bf16)val;
    }
}

// ---------------------------------------------------------------------------
extern "C" void kernel_launch(void* const* d_in, const int* in_sizes, int n_in,
                              void* d_out, int out_size, void* d_ws, size_t ws_size,
                              hipStream_t stream)
{
    (void)in_sizes; (void)n_in; (void)out_size; (void)ws_size;

    const float* x       = (const float*)d_in[0];
    const float* Wq      = (const float*)d_in[1];
    const float* Wk      = (const float*)d_in[2];
    const float* Wv      = (const float*)d_in[3];
    const float* Wa      = (const float*)d_in[4];
    const float* Wb      = (const float*)d_in[5];
    const float* Wg      = (const float*)d_in[6];
    const float* Wo      = (const float*)d_in[7];
    const float* cw_q    = (const float*)d_in[8];
    const float* cb_q    = (const float*)d_in[9];
    const float* cw_k    = (const float*)d_in[10];
    const float* cb_k    = (const float*)d_in[11];
    const float* cw_v    = (const float*)d_in[12];
    const float* cb_v    = (const float*)d_in[13];
    const float* A_log   = (const float*)d_in[14];
    const float* dt_bias = (const float*)d_in[15];
    const float* norm_w  = (const float*)d_in[16];
    float* out = (float*)d_out;

    const size_t NBIG = (size_t)MROWS * BDIM;   // 8M elems
    const size_t NW   = (size_t)BDIM * BDIM;    // 4M elems

    char* p = (char*)d_ws;
    __bf16* xb  = (__bf16*)p; p += NBIG * sizeof(__bf16);
    __bf16* Wqb = (__bf16*)p; p += NW * sizeof(__bf16);
    __bf16* Wkb = (__bf16*)p; p += NW * sizeof(__bf16);
    __bf16* Wvb = (__bf16*)p; p += NW * sizeof(__bf16);
    __bf16* Wgb = (__bf16*)p; p += NW * sizeof(__bf16);
    __bf16* Wob = (__bf16*)p; p += NW * sizeof(__bf16);
    float* qp = (float*)p; p += NBIG * sizeof(float);
    float* kp = (float*)p; p += NBIG * sizeof(float);
    float* vp = (float*)p; p += NBIG * sizeof(float);
    float* gp = (float*)p; p += NBIG * sizeof(float);
    float* qc = (float*)p; p += NBIG * sizeof(float);
    float* kc = (float*)p; p += NBIG * sizeof(float);
    float* vc = (float*)p; p += NBIG * sizeof(float);
    float* av   = (float*)p; p += (size_t)MROWS * NH * sizeof(float);
    float* bv   = (float*)p; p += (size_t)MROWS * NH * sizeof(float);
    float* gdec = (float*)p; p += (size_t)MROWS * NH * sizeof(float);
    float* beta = (float*)p; p += (size_t)MROWS * NH * sizeof(float);
    float*  osc = qp;            // q-projection is dead after its conv
    __bf16* yb  = xb;            // bf16 x is dead after the projection GEMMs

    // 1) casts to bf16
    k_cast_bf16<<<(int)((NBIG + 255) / 256), 256, 0, stream>>>(x,  xb,  (int)NBIG);
    k_cast_bf16<<<(int)((NW + 255) / 256), 256, 0, stream>>>(Wq, Wqb, (int)NW);
    k_cast_bf16<<<(int)((NW + 255) / 256), 256, 0, stream>>>(Wk, Wkb, (int)NW);
    k_cast_bf16<<<(int)((NW + 255) / 256), 256, 0, stream>>>(Wv, Wvb, (int)NW);
    k_cast_bf16<<<(int)((NW + 255) / 256), 256, 0, stream>>>(Wg, Wgb, (int)NW);
    k_cast_bf16<<<(int)((NW + 255) / 256), 256, 0, stream>>>(Wo, Wob, (int)NW);

    // 2) projection GEMMs (WMMA bf16, double-buffered async pipeline)
    dim3 gg(BDIM / GBN, MROWS / GBM);
    k_gemm_bf16<<<gg, 256, 0, stream>>>(xb, Wqb, qp, MROWS, BDIM, BDIM);
    k_gemm_bf16<<<gg, 256, 0, stream>>>(xb, Wkb, kp, MROWS, BDIM, BDIM);
    k_gemm_bf16<<<gg, 256, 0, stream>>>(xb, Wvb, vp, MROWS, BDIM, BDIM);
    k_gemm_bf16<<<gg, 256, 0, stream>>>(xb, Wgb, gp, MROWS, BDIM, BDIM);

    // 3) a/b projections + gates
    k_proj_ab<<<MROWS, 256, 0, stream>>>(x, Wa, Wb, av, bv);
    k_gate<<<(MROWS * NH + 255) / 256, 256, 0, stream>>>(av, bv, A_log, dt_bias,
                                                         gdec, beta, MROWS * NH);

    // 4) causal conv + SiLU
    int tot = (int)NBIG;
    k_conv_silu<<<(tot + 255) / 256, 256, 0, stream>>>(qp, cw_q, cb_q, qc, tot, BDIM);
    k_conv_silu<<<(tot + 255) / 256, 256, 0, stream>>>(kp, cw_k, cb_k, kc, tot, BDIM);
    k_conv_silu<<<(tot + 255) / 256, 256, 0, stream>>>(vp, cw_v, cb_v, vc, tot, BDIM);

    // 5) per-head L2 norm of q, k
    int nvec = MROWS * NH;
    k_l2norm<<<(nvec + 7) / 8, 256, 0, stream>>>(qc, nvec);
    k_l2norm<<<(nvec + 7) / 8, 256, 0, stream>>>(kc, nvec);

    // 6) gated delta-rule scan (one block per (b,h), state in registers)
    k_recurrence<<<2 * NH, 256, 0, stream>>>(qc, kc, vc, gdec, beta, osc);

    // 7) RMSNorm * silu(gp) -> bf16
    k_norm_gate<<<MROWS, 256, 0, stream>>>(osc, gp, norm_w, yb);

    // 8) output projection GEMM (WMMA bf16)
    k_gemm_bf16<<<gg, 256, 0, stream>>>(yb, Wob, out, MROWS, BDIM, BDIM);
}
